// GCMP_4879082848653
// MI455X (gfx1250) — matrix operand order repaired and checked by reading, not compiled, not run on hardware
//
#include <hip/hip_runtime.h>
#include <math.h>

#define N_NODES 100000
#define N_EDGES 1600000
#define NVC (N_NODES * 64)
#define NVE (N_EDGES * 64)
#define BN_EPS 1e-5f

typedef __bf16 bf16;
typedef bf16 v16bf __attribute__((ext_vector_type(16)));
typedef float v8f  __attribute__((ext_vector_type(8)));

static_assert(sizeof(v16bf) == 32, "v16bf must be 32 bytes");

// ---------------- helpers ----------------

__device__ __forceinline__ bf16 f2bf(float f) {
  unsigned u = __builtin_bit_cast(unsigned, f);
  u += 0x7FFFu + ((u >> 16) & 1u);          // round-to-nearest-even
  return __builtin_bit_cast(bf16, (unsigned short)(u >> 16));
}
__device__ __forceinline__ float bf2f(bf16 b) {
  unsigned u = ((unsigned)__builtin_bit_cast(unsigned short, b)) << 16;
  return __builtin_bit_cast(float, u);
}

// A-matrix (16x32 bf16) fragment slot for element (m row 0..15, k 0..31):
// lane = m | ((k&8)<<1) ; element j = ((k>>4)<<3) | (k&7)   (ISA 7.12.2)
__device__ __forceinline__ int a_slot(int m, int k) {
  return ((m | ((k & 8) << 1)) << 4) | (((k >> 4) << 3) | (k & 7));
}

// load one lane's 16 bf16 fragment (32B) assuming only 16B alignment
__device__ __forceinline__ v16bf load_frag(const bf16* p) {
  struct alignas(16) H { float4 a, b; };
  const float4* q = (const float4*)p;
  H h; h.a = q[0]; h.b = q[1];
  return __builtin_bit_cast(v16bf, h);
}

__device__ __forceinline__ v8f wmma_step(const bf16* ap, const bf16* bp, v8f c) {
  v16bf a = load_frag(ap);
  v16bf b = load_frag(bp);
  return __builtin_amdgcn_wmma_f32_16x16x32_bf16(false, a, false, b, (short)0, c,
                                                 false, false);
}

// monotone-bits float atomic max/min (s2 init -inf, s3 init +inf)
__device__ __forceinline__ void atomicMaxF(float* a, float v) {
  if (v >= 0.f) atomicMax((int*)a, __float_as_int(v));
  else          atomicMin((unsigned int*)a, __float_as_uint(v));
}
__device__ __forceinline__ void atomicMinF(float* a, float v) {
  if (v >= 0.f) atomicMin((int*)a, __float_as_int(v));
  else          atomicMax((unsigned int*)a, __float_as_uint(v));
}

// ---------------- weight pre-pack (fp32 -> bf16, B-fragment order) ----------------
// pack[(nt*kt_cnt + kt)*512 + lane*16 + j] ; lane = n | ((k&8)<<1) ; j = ((k>>4)<<3) | (k&7)
__global__ void gcmp_pack_b(const float* __restrict__ W, int ldw, int col0,
                            int kt_cnt, int total, bf16* __restrict__ out) {
  for (int idx = blockIdx.x * blockDim.x + threadIdx.x; idx < total;
       idx += gridDim.x * blockDim.x) {
    int slot = idx & 511;
    int tile = idx >> 9;
    int kt = tile % kt_cnt;
    int nt = tile / kt_cnt;
    int lane = slot >> 4, j = slot & 15;
    int k = (j & 7) | ((lane & 16) >> 1) | ((j & 8) << 1);
    int n = lane & 15;
    out[idx] = f2bf(W[(size_t)(kt * 32 + k) * ldw + col0 + nt * 16 + n]);
  }
}

__global__ void gcmp_pack_gatecol(const float* __restrict__ W2,
                                  float* __restrict__ w2k) {
  int i = blockIdx.x * blockDim.x + threadIdx.x;
  if (i < 192) w2k[i] = W2[(size_t)i * 257];   // W_msg2[:, 0]
}

// ---------------- init scatter buffers + BN accumulators ----------------
__global__ void gcmp_init(float* __restrict__ s1, float* __restrict__ s2,
                          float* __restrict__ s3, float* __restrict__ s4,
                          float* __restrict__ deg, float* __restrict__ bn) {
  const float NEGINF = __int_as_float(0xFF800000);
  const float POSINF = __int_as_float(0x7F800000);
  int stride = gridDim.x * blockDim.x;
  for (int i = blockIdx.x * blockDim.x + threadIdx.x; i < NVC; i += stride) {
    s1[i] = 0.f; s2[i] = NEGINF; s3[i] = POSINF; s4[i] = 0.f;
    if (i < N_NODES) deg[i] = 0.f;
    if (i < 1024) bn[i] = 0.f;
  }
}

// ---------------- fused edge pipeline ----------------
// per 64-edge tile: gather -> GEMM1(relu) -> gate -> GEMM2(*gate) ->
// scatter atomics to s1..s4 -> GEMM3 -> pre-BN out_ve + BN stats
__global__ __launch_bounds__(256) void gcmp_edge_kernel(
    const float* __restrict__ in_vc, const float* __restrict__ in_ve,
    const int* __restrict__ src, const int* __restrict__ dst,
    const float* __restrict__ b_msg1, const float* __restrict__ b_msg2,
    const float* __restrict__ b_ve, const bf16* __restrict__ W1p,
    const bf16* __restrict__ W2p, const bf16* __restrict__ Wvp,
    const float* __restrict__ w2k, float* __restrict__ s1,
    float* __restrict__ s2, float* __restrict__ s3, float* __restrict__ s4,
    float* __restrict__ deg, float* __restrict__ bn,
    float* __restrict__ out_ve) {
  extern __shared__ char smem[];
  bf16* xf = (bf16*)smem;        // [4 mt][6 kt][512]  x  (24576 B)
  bf16* hf = xf + 12288;         // [4][6][512]        h  (24576 B)
  bf16* ff = hf + 12288;         // [4][8][512]        f  (32768 B)
  float* lb1  = (float*)(ff + 16384);  // 192
  float* lb2  = lb1 + 192;             // 256 (b_msg2[1..256])
  float* lw2k = lb2 + 256;             // 192 (gate column)
  float* lbve = lw2k + 192;            // 64
  float* gate = lbve + 64;             // 64
  float* bnS  = gate + 64;             // 64
  float* bnQ  = bnS + 64;              // 64
  int* sIdx = (int*)(bnQ + 64);        // 64
  int* dIdx = sIdx + 64;               // 64

  const int tid  = threadIdx.x;
  const int lane = tid & 31;
  const int wave = tid >> 5;

  for (int i = tid; i < 192; i += 256) { lb1[i] = b_msg1[i]; lw2k[i] = w2k[i]; }
  if (tid < 256) lb2[tid] = b_msg2[1 + tid];
  if (tid < 64) { lbve[tid] = b_ve[tid]; bnS[tid] = 0.f; bnQ[tid] = 0.f; }
  const float bgate = b_msg2[0];
  __syncthreads();

  for (int t = blockIdx.x; t < N_EDGES / 64; t += gridDim.x) {
    if (tid < 64) {
      int e = t * 64 + tid;
      int sv = src[e], dv = dst[e];
      sIdx[tid] = sv; dIdx[tid] = dv;
      gate[tid] = bgate;
      atomicAdd(&deg[dv], 1.0f);
      __builtin_prefetch(src + (size_t)(t + gridDim.x) * 64 + tid, 0, 0);
      __builtin_prefetch(dst + (size_t)(t + gridDim.x) * 64 + tid, 0, 0);
    }
    __syncthreads();

    // gather x = [vc[src] | vc[dst] | ve] into A-fragment layout (bf16)
    for (int i = tid; i < 64 * 48; i += 256) {
      int e = i / 48, q = i - e * 48;
      int seg = q >> 4, q4 = q & 15;
      const float4* rowp;
      if (seg == 0)      rowp = (const float4*)(in_vc + (size_t)sIdx[e] * 64);
      else if (seg == 1) rowp = (const float4*)(in_vc + (size_t)dIdx[e] * 64);
      else               rowp = (const float4*)(in_ve + ((size_t)t * 64 + e) * 64);
      float4 v = rowp[q4];
      float vv[4] = {v.x, v.y, v.z, v.w};
      int fb = seg * 64 + q4 * 4;
#pragma unroll
      for (int c2 = 0; c2 < 4; ++c2) {
        int f = fb + c2;
        xf[((e >> 4) * 6 + (f >> 5)) * 512 + a_slot(e & 15, f & 31)] = f2bf(vv[c2]);
      }
    }
    __syncthreads();

    // GEMM1: h = relu(x @ W1 + b1)   M=64 N=192 K=192 -> 48 tiles / 8 waves
    for (int tt = wave; tt < 48; tt += 8) {
      int mt = tt / 12, nt = tt - mt * 12;
      float bias = lb1[nt * 16 + (lane & 15)];
      v8f c;
#pragma unroll
      for (int r = 0; r < 8; ++r) c[r] = bias;
      const bf16* ap = xf + mt * (6 * 512) + (lane << 4);
      const bf16* bp = W1p + (nt * 6) * 512 + (lane << 4);
#pragma unroll
      for (int kt = 0; kt < 6; ++kt) c = wmma_step(ap + kt * 512, bp + kt * 512, c);
      int col = nt * 16 + (lane & 15);
      int mhi = (lane >> 4) << 3;
#pragma unroll
      for (int r = 0; r < 8; ++r) {
        float v = c[r] > 0.f ? c[r] : 0.f;
        hf[(mt * 6 + (col >> 5)) * 512 + a_slot(mhi + r, col & 31)] = f2bf(v);
      }
    }
    __syncthreads();

    // gate = sigmoid(h @ W_msg2[:,0] + b_msg2[0]) ; 4 threads per edge
    {
      int e = tid >> 2, p = tid & 3;
      float s = 0.f;
      int f0 = p * 48;
      for (int f = f0; f < f0 + 48; ++f)
        s += bf2f(hf[((e >> 4) * 6 + (f >> 5)) * 512 + a_slot(e & 15, f & 31)]) *
             lw2k[f];
      atomicAdd(&gate[e], s);
    }
    __syncthreads();
    if (tid < 64) gate[tid] = 1.f / (1.f + expf(-gate[tid]));
    __syncthreads();

    // GEMM2: m = h @ W2[:,1:257] + b2 ; f = m * gate ; scatter + stage for GEMM3
    for (int tt = wave; tt < 64; tt += 8) {
      int mt = tt >> 4, nt = tt & 15;
      float bias = lb2[nt * 16 + (lane & 15)];
      v8f c;
#pragma unroll
      for (int r = 0; r < 8; ++r) c[r] = bias;
      const bf16* ap = hf + mt * (6 * 512) + (lane << 4);
      const bf16* bp = W2p + (nt * 6) * 512 + (lane << 4);
#pragma unroll
      for (int kt = 0; kt < 6; ++kt) c = wmma_step(ap + kt * 512, bp + kt * 512, c);
      int col = nt * 16 + (lane & 15);          // 0..255 -> f1..f4
      int grp = col >> 6, cc = col & 63;
      int mhi = (lane >> 4) << 3;
#pragma unroll
      for (int r = 0; r < 8; ++r) {
        int mrow = mhi + r;
        int e = mt * 16 + mrow;
        float fv = c[r] * gate[e];
        int node = dIdx[e];
        if (grp == 0)      atomicAdd(&s1[(size_t)node * 64 + cc], fv);
        else if (grp == 1) atomicMaxF(&s2[(size_t)node * 64 + cc], fv);
        else if (grp == 2) atomicMinF(&s3[(size_t)node * 64 + cc], fv);
        else               atomicAdd(&s4[(size_t)node * 64 + cc], fv);
        ff[(mt * 8 + (col >> 5)) * 512 + a_slot(mrow, col & 31)] = f2bf(fv);
      }
    }
    __syncthreads();

    // GEMM3: out_ve = [f1 f2 f3 f4 | in_ve] @ W_ve + b_ve  (K=320 -> 10 ktiles)
    for (int tt = wave; tt < 16; tt += 8) {
      int mt = tt >> 2, nt = tt & 3;
      float bias = lbve[nt * 16 + (lane & 15)];
      v8f c;
#pragma unroll
      for (int r = 0; r < 8; ++r) c[r] = bias;
#pragma unroll
      for (int kt = 0; kt < 10; ++kt) {
        const bf16* ap = (kt < 8) ? (ff + (mt * 8 + kt) * 512)
                                  : (xf + (mt * 6 + 4 + (kt - 8)) * 512);
        c = wmma_step(ap + (lane << 4), Wvp + (nt * 10 + kt) * 512 + (lane << 4), c);
      }
      int col = nt * 16 + (lane & 15);
      int mhi = (lane >> 4) << 3;
#pragma unroll
      for (int r = 0; r < 8; ++r) {
        int e = mt * 16 + mhi + r;
        size_t ge = (size_t)t * 64 + e;
        float v = c[r];
        out_ve[ge * 64 + col] = v;      // pre-BN, normalized later in place
        atomicAdd(&bnS[col], v);
        atomicAdd(&bnQ[col], v * v);
      }
    }
    __syncthreads();
  }

  if (tid < 64) {
    atomicAdd(&bn[128 + tid], bnS[tid]);
    atomicAdd(&bn[192 + tid], bnQ[tid]);
  }
}

// ---------------- node pipeline ----------------
__global__ __launch_bounds__(256) void gcmp_node_kernel(
    const float* __restrict__ in_vc, const float* __restrict__ s1,
    const float* __restrict__ s2, const float* __restrict__ s3,
    const float* __restrict__ s4, const float* __restrict__ deg,
    const bf16* __restrict__ Wrp, const float* __restrict__ b_red,
    float* __restrict__ out, float* __restrict__ bn) {
  extern __shared__ char smem[];
  bf16* af = (bf16*)smem;              // [4][10][512] = 40960 B
  float* lbr  = (float*)(af + 20480);  // 64
  float* degv = lbr + 64;              // 64
  float* bnS  = degv + 64;             // 64
  float* bnQ  = bnS + 64;              // 64

  const int tid  = threadIdx.x;
  const int lane = tid & 31;
  const int wave = tid >> 5;

  if (tid < 64) { lbr[tid] = b_red[tid]; bnS[tid] = 0.f; bnQ[tid] = 0.f; }
  __syncthreads();

  const int NTN = (N_NODES + 63) / 64;
  for (int t = blockIdx.x; t < NTN; t += gridDim.x) {
    if (tid < 64) {
      int g = t * 64 + tid;
      degv[tid] = (g < N_NODES) ? deg[g] : 0.f;
    }
    __syncthreads();

    // gather [in_vc | s1 | fix(s2) | fix(s3) | s4/deg] -> A fragments
    for (int i = tid; i < 64 * 80; i += 256) {
      int e = i / 80, q = i - e * 80;
      int seg = q >> 4, q4 = q & 15;
      int g = t * 64 + e;
      float vv[4] = {0.f, 0.f, 0.f, 0.f};
      if (g < N_NODES) {
        const float* base = (seg == 0) ? in_vc
                          : (seg == 1) ? s1
                          : (seg == 2) ? s2
                          : (seg == 3) ? s3 : s4;
        float4 v = ((const float4*)(base + (size_t)g * 64))[q4];
        vv[0] = v.x; vv[1] = v.y; vv[2] = v.z; vv[3] = v.w;
        float d = degv[e];
        if (seg == 2 || seg == 3) {
          if (!(d > 0.f)) { vv[0] = vv[1] = vv[2] = vv[3] = 0.f; }
        } else if (seg == 4) {
          float inv = 1.f / fmaxf(d, 1.f);
          vv[0] *= inv; vv[1] *= inv; vv[2] *= inv; vv[3] *= inv;
        }
      }
      int fb = seg * 64 + q4 * 4;
#pragma unroll
      for (int c2 = 0; c2 < 4; ++c2) {
        int f = fb + c2;
        af[((e >> 4) * 10 + (f >> 5)) * 512 + a_slot(e & 15, f & 31)] = f2bf(vv[c2]);
      }
    }
    __syncthreads();

    // GEMM: out_vc = A @ W_red + b_red   M=64 N=64 K=320 -> 16 tiles / 8 waves
    for (int tt = wave; tt < 16; tt += 8) {
      int mt = tt >> 2, nt = tt & 3;
      float bias = lbr[nt * 16 + (lane & 15)];
      v8f c;
#pragma unroll
      for (int r = 0; r < 8; ++r) c[r] = bias;
      const bf16* ap = af + mt * (10 * 512) + (lane << 4);
      const bf16* bp = Wrp + (nt * 10) * 512 + (lane << 4);
#pragma unroll
      for (int kt = 0; kt < 10; ++kt) c = wmma_step(ap + kt * 512, bp + kt * 512, c);
      int col = nt * 16 + (lane & 15);
      int mhi = (lane >> 4) << 3;
#pragma unroll
      for (int r = 0; r < 8; ++r) {
        int g = t * 64 + mt * 16 + mhi + r;
        if (g < N_NODES) {
          float v = c[r];
          out[(size_t)g * 64 + col] = v;
          atomicAdd(&bnS[col], v);
          atomicAdd(&bnQ[col], v * v);
        }
      }
    }
    __syncthreads();
  }

  if (tid < 64) {
    atomicAdd(&bn[tid], bnS[tid]);
    atomicAdd(&bn[64 + tid], bnQ[tid]);
  }
}

// ---------------- batchnorm finalize + apply ----------------
__global__ void gcmp_bn_finalize(float* __restrict__ bn,
                                 const float* __restrict__ g_vc,
                                 const float* __restrict__ b_vc,
                                 const float* __restrict__ g_ve,
                                 const float* __restrict__ b_ve) {
  int tid = threadIdx.x;
  if (tid < 64) {
    float mu = bn[tid] / (float)N_NODES;
    float var = bn[64 + tid] / (float)N_NODES - mu * mu;
    bn[256 + tid] = mu;
    bn[320 + tid] = g_vc[tid] * rsqrtf(var + BN_EPS);
    bn[384 + tid] = b_vc[tid];
  } else if (tid < 128) {
    int c = tid - 64;
    float mu = bn[128 + c] / (float)N_EDGES;
    float var = bn[192 + c] / (float)N_EDGES - mu * mu;
    bn[448 + c] = mu;
    bn[512 + c] = g_ve[c] * rsqrtf(var + BN_EPS);
    bn[576 + c] = b_ve[c];
  }
}

__global__ void gcmp_bn_apply(float* __restrict__ out,
                              const float* __restrict__ bn) {
  const int NV4 = NVC / 4;                 // vc/ve boundary in float4 units
  const int TOT4 = (NVC + NVE) / 4;
  float4* o4 = (float4*)out;
  int stride = gridDim.x * blockDim.x;
  for (int i = blockIdx.x * blockDim.x + threadIdx.x; i < TOT4; i += stride) {
    int base = (i < NV4) ? 256 : 448;
    int c0 = (i & 15) * 4;
    float4 v = o4[i];
    v.x = (v.x - bn[base + c0 + 0]) * bn[base + 64 + c0 + 0] + bn[base + 128 + c0 + 0];
    v.y = (v.y - bn[base + c0 + 1]) * bn[base + 64 + c0 + 1] + bn[base + 128 + c0 + 1];
    v.z = (v.z - bn[base + c0 + 2]) * bn[base + 64 + c0 + 2] + bn[base + 128 + c0 + 2];
    v.w = (v.w - bn[base + c0 + 3]) * bn[base + 64 + c0 + 3] + bn[base + 128 + c0 + 3];
    o4[i] = v;
  }
}

// ---------------- host launcher ----------------
extern "C" void kernel_launch(void* const* d_in, const int* in_sizes, int n_in,
                              void* d_out, int out_size, void* d_ws,
                              size_t ws_size, hipStream_t stream) {
  (void)in_sizes; (void)n_in; (void)out_size; (void)ws_size;
  const float* in_vc    = (const float*)d_in[0];
  const float* in_ve    = (const float*)d_in[1];
  const int*   src      = (const int*)d_in[2];
  const int*   dst      = (const int*)d_in[3];
  const float* W_msg1   = (const float*)d_in[4];
  const float* b_msg1   = (const float*)d_in[5];
  const float* W_msg2   = (const float*)d_in[6];
  const float* b_msg2   = (const float*)d_in[7];
  const float* W_red    = (const float*)d_in[8];
  const float* b_red    = (const float*)d_in[9];
  const float* W_ve     = (const float*)d_in[10];
  const float* b_ve     = (const float*)d_in[11];
  const float* gamma_gc = (const float*)d_in[12];
  const float* beta_gc  = (const float*)d_in[13];
  const float* gamma_ef = (const float*)d_in[14];
  const float* beta_ef  = (const float*)d_in[15];
  float* out = (float*)d_out;

  char* ws = (char*)d_ws;
  size_t off = 0;
  auto alloc = [&](size_t bytes) {
    void* p = ws + off;
    off = (off + bytes + 255) & ~(size_t)255;
    return p;
  };
  bf16* W1p  = (bf16*)alloc((size_t)12 * 6 * 512 * sizeof(bf16));
  bf16* W2p  = (bf16*)alloc((size_t)16 * 6 * 512 * sizeof(bf16));
  bf16* Wvp  = (bf16*)alloc((size_t)4 * 10 * 512 * sizeof(bf16));
  bf16* Wrp  = (bf16*)alloc((size_t)4 * 10 * 512 * sizeof(bf16));
  float* w2k = (float*)alloc(192 * sizeof(float));
  float* s1  = (float*)alloc((size_t)NVC * sizeof(float));
  float* s2  = (float*)alloc((size_t)NVC * sizeof(float));
  float* s3  = (float*)alloc((size_t)NVC * sizeof(float));
  float* s4  = (float*)alloc((size_t)NVC * sizeof(float));
  float* deg = (float*)alloc((size_t)N_NODES * sizeof(float));
  float* bn  = (float*)alloc(1024 * sizeof(float));

  // 1) pack weights into bf16 WMMA-B fragment order
  gcmp_pack_b<<<144, 256, 0, stream>>>(W_msg1, 192, 0, 6, 12 * 6 * 512, W1p);
  gcmp_pack_b<<<192, 256, 0, stream>>>(W_msg2, 257, 1, 6, 16 * 6 * 512, W2p);
  gcmp_pack_b<<<80, 256, 0, stream>>>(W_ve, 64, 0, 10, 4 * 10 * 512, Wvp);
  gcmp_pack_b<<<80, 256, 0, stream>>>(W_red, 64, 0, 10, 4 * 10 * 512, Wrp);
  gcmp_pack_gatecol<<<1, 256, 0, stream>>>(W_msg2, w2k);

  // 2) init scatter buffers and BN accumulators
  gcmp_init<<<2048, 256, 0, stream>>>(s1, s2, s3, s4, deg, bn);

  // 3) fused edge pipeline (gather -> WMMA MLPs -> scatter -> out_ve pre-BN)
  const int SMEM_E = (12288 + 12288 + 16384) * 2 /*bf16*/ +
                     (192 + 256 + 192 + 64 + 64 + 64 + 64) * 4 /*f32*/ +
                     128 * 4 /*int*/;          // 86016 B (dynamic: WGP has 320KB)
  gcmp_edge_kernel<<<2048, 256, SMEM_E, stream>>>(
      in_vc, in_ve, src, dst, b_msg1, b_msg2, b_ve, W1p, W2p, Wvp, w2k, s1, s2,
      s3, s4, deg, bn, out + (size_t)NVC);

  // 4) node pipeline (fixups + WMMA GEMM -> out_vc pre-BN)
  const int SMEM_N = 20480 * 2 + 4 * 64 * 4;   // 41984 B
  gcmp_node_kernel<<<1563, 256, SMEM_N, stream>>>(in_vc, s1, s2, s3, s4, deg,
                                                  Wrp, b_red, out, bn);

  // 5) batch-norm stats -> affine params, then in-place normalize d_out
  gcmp_bn_finalize<<<1, 128, 0, stream>>>(bn, gamma_gc, beta_gc, gamma_ef, beta_ef);
  gcmp_bn_apply<<<4096, 256, 0, stream>>>(out, bn);
}
